// Conv_82506321756838
// MI455X (gfx1250) — matrix operand order repaired and checked by reading, not compiled, not run on hardware
//
#include <hip/hip_runtime.h>
#include <math.h>

#define Hdim 128

typedef __attribute__((ext_vector_type(16))) _Float16 v16h;
typedef __attribute__((ext_vector_type(8)))  float    v8f;

__device__ __forceinline__ float gelu_f(float x) {
  return 0.5f * x * (1.0f + erff(x * 0.70710678118654752f));
}

__device__ __forceinline__ v8f wmma_f16(v16h a, v16h b, v8f c) {
  return __builtin_amdgcn_wmma_f32_16x16x32_f16(false, a, false, b, (short)0, c, false, false);
}

// ---------- prep: weights -> transposed f16 hi/lo planes; zero BN stats ----------
__global__ __launch_bounds__(256) void prep_kernel(
    const float* __restrict__ preW, const float* __restrict__ W1,
    const float* __restrict__ W2, _Float16* __restrict__ whl,
    float* __restrict__ stats) {
  int idx = blockIdx.x * 256 + threadIdx.x;
  if (idx < 512) stats[idx] = 0.0f;
  if (idx < 3 * Hdim * Hdim) {
    int w = idx / (Hdim * Hdim);
    int rem = idx - w * (Hdim * Hdim);
    int k = rem / Hdim;
    int n = rem - k * Hdim;
    const float* Wsrc = (w == 0) ? preW : (w == 1) ? W1 : W2;
    float f = Wsrc[k * Hdim + n];
    _Float16 h = (_Float16)f;
    _Float16 l = (_Float16)(f - (float)h);
    _Float16* base = whl + (size_t)w * 2 * Hdim * Hdim;
    base[n * Hdim + k] = h;                    // hi plane, transposed [n][k]
    base[Hdim * Hdim + n * Hdim + k] = l;      // lo plane, transposed
  }
}

// ---------- copy x_feat -> x_buf (scatter target) ----------
__global__ __launch_bounds__(256) void copy_kernel(const float4* __restrict__ in,
                                                   float4* __restrict__ out, int n4) {
  int idx = blockIdx.x * 256 + threadIdx.x;
  if (idx < n4) out[idx] = in[idx];
}

// ---------- edge path: gelu((x[src]+eattr)@preW + pre_b)*bases, atomic scatter ----------
__global__ __launch_bounds__(256) void edge_kernel(
    const float* __restrict__ x_feat, const float* __restrict__ edge_attr,
    const float* __restrict__ bases, const int* __restrict__ src,
    const int* __restrict__ dst, const float* __restrict__ pre_b,
    const _Float16* __restrict__ wThi, const _Float16* __restrict__ wTlo,
    float* __restrict__ x_out, int E) {
  __shared__ __align__(32) _Float16 sWhi[Hdim * Hdim];
  __shared__ __align__(32) _Float16 sWlo[Hdim * Hdim];
  int tid = threadIdx.x;

  // stage 2 x 32KB weight planes into LDS (2048 uint4 each)
  {
    const uint4* ghi = (const uint4*)wThi;
    const uint4* glo = (const uint4*)wTlo;
    uint4* shi = (uint4*)sWhi;
    uint4* slo = (uint4*)sWlo;
#pragma unroll
    for (int i = 0; i < 8; ++i) {
      shi[tid + i * 256] = ghi[tid + i * 256];
      slo[tid + i * 256] = glo[tid + i * 256];
    }
  }
  __syncthreads();

  int wave = tid >> 5;
  int lane = tid & 31;
  int g = lane >> 4;       // half-wave group
  int r = lane & 15;       // A row within tile / output column within N-tile
  int tileBase = blockIdx.x * 128 + wave * 16;
  if (tileBase >= E) return;

  int myRow = tileBase + r;
  if (myRow >= E) myRow = E - 1;
  int s = src[myRow];
  const float* xrow = x_feat + (size_t)s * Hdim;
  const float* erow = edge_attr + (size_t)myRow * Hdim;

  v8f acc[8];
#pragma unroll
  for (int t = 0; t < 8; ++t)
#pragma unroll
    for (int j = 0; j < 8; ++j) acc[t][j] = 0.0f;

#pragma unroll
  for (int k0 = 0; k0 < Hdim; k0 += 32) {
    int kA = k0 + g * 16;
    float4 xa = *(const float4*)(xrow + kA);
    float4 xb = *(const float4*)(xrow + kA + 4);
    float4 xc = *(const float4*)(xrow + kA + 8);
    float4 xd = *(const float4*)(xrow + kA + 12);
    float4 ea = *(const float4*)(erow + kA);
    float4 eb = *(const float4*)(erow + kA + 4);
    float4 ec = *(const float4*)(erow + kA + 8);
    float4 ed = *(const float4*)(erow + kA + 12);
    float va[16] = {xa.x + ea.x, xa.y + ea.y, xa.z + ea.z, xa.w + ea.w,
                    xb.x + eb.x, xb.y + eb.y, xb.z + eb.z, xb.w + eb.w,
                    xc.x + ec.x, xc.y + ec.y, xc.z + ec.z, xc.w + ec.w,
                    xd.x + ed.x, xd.y + ed.y, xd.z + ed.z, xd.w + ed.w};
    v16h ah, al;
#pragma unroll
    for (int e = 0; e < 16; ++e) {
      _Float16 h = (_Float16)va[e];
      ah[e] = h;
      al[e] = (_Float16)(va[e] - (float)h);
    }
#pragma unroll
    for (int t = 0; t < 8; ++t) {
      int boff = (t * 16 + r) * Hdim + k0 + g * 16;
      v16h bh = *(const v16h*)(sWhi + boff);
      v16h bl = *(const v16h*)(sWlo + boff);
      acc[t] = wmma_f16(ah, bh, acc[t]);
      acc[t] = wmma_f16(ah, bl, acc[t]);
      acc[t] = wmma_f16(al, bh, acc[t]);
    }
  }

  float pb[8];
#pragma unroll
  for (int t = 0; t < 8; ++t) pb[t] = pre_b[t * 16 + r];

#pragma unroll
  for (int v = 0; v < 8; ++v) {
    int e = tileBase + v + g * 8;
    if (e < E) {
      int d = dst[e];
      const float* brow = bases + (size_t)e * Hdim;
      float* orow = x_out + (size_t)d * Hdim;
#pragma unroll
      for (int t = 0; t < 8; ++t) {
        int n = t * 16 + r;
        float val = acc[t][v] + pb[t];
        val = gelu_f(val) * brow[n];
        __hip_atomic_fetch_add(orow + n, val, __ATOMIC_RELAXED, __HIP_MEMORY_SCOPE_AGENT);
      }
    }
  }
}

// ---------- node GEMM (+ optional input BN+GELU) + BN-stat reduction ----------
template <bool APPLY_BN>
__global__ __launch_bounds__(256) void node_gemm_kernel(
    const float* __restrict__ A,
    const _Float16* __restrict__ wThi, const _Float16* __restrict__ wTlo,
    const float* __restrict__ bias,
    const float* __restrict__ inSum, const float* __restrict__ inSq,
    const float* __restrict__ gamma, const float* __restrict__ betav,
    float* __restrict__ Tout, float* __restrict__ outSum, float* __restrict__ outSq,
    int Nrows, float invN) {
  __shared__ float sA[Hdim], sC[Hdim], sSum[Hdim], sSq[Hdim];
  int tid = threadIdx.x;
  if (tid < Hdim) {
    sSum[tid] = 0.0f;
    sSq[tid] = 0.0f;
    if (APPLY_BN) {
      float mu = inSum[tid] * invN;
      float var = inSq[tid] * invN - mu * mu;
      float a = gamma[tid] * rsqrtf(var + 1e-5f);
      sA[tid] = a;
      sC[tid] = betav[tid] - a * mu;
    }
  }
  __syncthreads();

  int wave = tid >> 5, lane = tid & 31;
  int g = lane >> 4, r = lane & 15;
  int tileBase = blockIdx.x * 128 + wave * 16;

  if (tileBase < Nrows) {
    int myRow = tileBase + r;
    if (myRow >= Nrows) myRow = Nrows - 1;
    const float* arow = A + (size_t)myRow * Hdim;

    v8f acc[8];
#pragma unroll
    for (int t = 0; t < 8; ++t)
#pragma unroll
      for (int j = 0; j < 8; ++j) acc[t][j] = 0.0f;

#pragma unroll
    for (int k0 = 0; k0 < Hdim; k0 += 32) {
      int kA = k0 + g * 16;
      float4 xa = *(const float4*)(arow + kA);
      float4 xb = *(const float4*)(arow + kA + 4);
      float4 xc = *(const float4*)(arow + kA + 8);
      float4 xd = *(const float4*)(arow + kA + 12);
      float va[16] = {xa.x, xa.y, xa.z, xa.w, xb.x, xb.y, xb.z, xb.w,
                      xc.x, xc.y, xc.z, xc.w, xd.x, xd.y, xd.z, xd.w};
      v16h ah, al;
#pragma unroll
      for (int e = 0; e < 16; ++e) {
        float f = va[e];
        if (APPLY_BN) {
          int k = kA + e;  // contiguous A-fragment K mapping
          f = gelu_f(sA[k] * f + sC[k]);
        }
        _Float16 h = (_Float16)f;
        ah[e] = h;
        al[e] = (_Float16)(f - (float)h);
      }
#pragma unroll
      for (int t = 0; t < 8; ++t) {
        int boff = (t * 16 + r) * Hdim + k0 + g * 16;
        v16h bh = *(const v16h*)(wThi + boff);
        v16h bl = *(const v16h*)(wTlo + boff);
        acc[t] = wmma_f16(ah, bh, acc[t]);
        acc[t] = wmma_f16(ah, bl, acc[t]);
        acc[t] = wmma_f16(al, bh, acc[t]);
      }
    }

#pragma unroll
    for (int t = 0; t < 8; ++t) {
      int n = t * 16 + r;
      float bsn = bias[n];
      float psum = 0.0f, psq = 0.0f;
#pragma unroll
      for (int v = 0; v < 8; ++v) {
        int row = tileBase + v + g * 8;
        if (row < Nrows) {
          float val = acc[t][v] + bsn;
          Tout[(size_t)row * Hdim + n] = val;
          psum += val;
          psq += val * val;
        }
      }
      __hip_atomic_fetch_add(&sSum[n], psum, __ATOMIC_RELAXED, __HIP_MEMORY_SCOPE_WORKGROUP);
      __hip_atomic_fetch_add(&sSq[n], psq, __ATOMIC_RELAXED, __HIP_MEMORY_SCOPE_WORKGROUP);
    }
  }
  __syncthreads();
  if (tid < Hdim) {
    __hip_atomic_fetch_add(&outSum[tid], sSum[tid], __ATOMIC_RELAXED, __HIP_MEMORY_SCOPE_AGENT);
    __hip_atomic_fetch_add(&outSq[tid], sSq[tid], __ATOMIC_RELAXED, __HIP_MEMORY_SCOPE_AGENT);
  }
}

// ---------- final: out = x + gelu(bn2(t2)) ----------
__global__ __launch_bounds__(256) void final_kernel(
    const float* __restrict__ x_buf, const float* __restrict__ t2,
    const float* __restrict__ sum2, const float* __restrict__ sq2,
    const float* __restrict__ g2, const float* __restrict__ beta2,
    float* __restrict__ out, int total, float invN) {
  int idx = (blockIdx.x * 256 + threadIdx.x) * 4;
  if (idx >= total) return;
  float4 tv = *(const float4*)(t2 + idx);
  float4 xv = *(const float4*)(x_buf + idx);
  float tvv[4] = {tv.x, tv.y, tv.z, tv.w};
  float xvv[4] = {xv.x, xv.y, xv.z, xv.w};
  float res[4];
#pragma unroll
  for (int j = 0; j < 4; ++j) {
    int col = (idx + j) & (Hdim - 1);
    float mu = sum2[col] * invN;
    float var = sq2[col] * invN - mu * mu;
    float a = g2[col] * rsqrtf(var + 1e-5f);
    float c = beta2[col] - a * mu;
    res[j] = xvv[j] + gelu_f(a * tvv[j] + c);
  }
  *(float4*)(out + idx) = make_float4(res[0], res[1], res[2], res[3]);
}

extern "C" void kernel_launch(void* const* d_in, const int* in_sizes, int n_in,
                              void* d_out, int out_size, void* d_ws, size_t ws_size,
                              hipStream_t stream) {
  const float* x_feat    = (const float*)d_in[0];
  const float* edge_attr = (const float*)d_in[1];
  const float* bases     = (const float*)d_in[2];
  const int*   src       = (const int*)d_in[3];
  const int*   dst       = (const int*)d_in[4];
  const float* pre_W     = (const float*)d_in[5];
  const float* pre_b     = (const float*)d_in[6];
  const float* W1        = (const float*)d_in[7];
  const float* b1        = (const float*)d_in[8];
  const float* g1        = (const float*)d_in[9];
  const float* beta1     = (const float*)d_in[10];
  const float* W2        = (const float*)d_in[11];
  const float* b2        = (const float*)d_in[12];
  const float* g2        = (const float*)d_in[13];
  const float* beta2     = (const float*)d_in[14];

  int N = in_sizes[0] / Hdim;
  int E = in_sizes[3];
  size_t NH = (size_t)N * Hdim;
  float invN = 1.0f / (float)N;

  // workspace layout
  float* x_buf = (float*)d_ws;               // N*H
  float* t1    = x_buf + NH;                 // N*H
  float* t2    = t1 + NH;                    // N*H
  float* stats = t2 + NH;                    // 512: sum1,sq1,sum2,sq2
  _Float16* whl = (_Float16*)(stats + 512);  // 6 planes of H*H f16
  _Float16* preWT = whl;
  _Float16* W1T   = whl + 2 * Hdim * Hdim;
  _Float16* W2T   = whl + 4 * Hdim * Hdim;

  prep_kernel<<<(3 * Hdim * Hdim + 255) / 256, 256, 0, stream>>>(pre_W, W1, W2, whl, stats);
  copy_kernel<<<((int)(NH / 4) + 255) / 256, 256, 0, stream>>>(
      (const float4*)x_feat, (float4*)x_buf, (int)(NH / 4));
  edge_kernel<<<(E + 127) / 128, 256, 0, stream>>>(
      x_feat, edge_attr, bases, src, dst, pre_b,
      preWT, preWT + Hdim * Hdim, x_buf, E);
  int nodeBlocks = (N + 127) / 128;
  node_gemm_kernel<false><<<nodeBlocks, 256, 0, stream>>>(
      x_buf, W1T, W1T + Hdim * Hdim, b1,
      nullptr, nullptr, nullptr, nullptr,
      t1, stats, stats + 128, N, invN);
  node_gemm_kernel<true><<<nodeBlocks, 256, 0, stream>>>(
      t1, W2T, W2T + Hdim * Hdim, b2,
      stats, stats + 128, g1, beta1,
      t2, stats + 256, stats + 384, N, invN);
  final_kernel<<<((int)(NH / 4) + 255) / 256, 256, 0, stream>>>(
      x_buf, t2, stats + 256, stats + 384, g2, beta2, (float*)d_out, (int)NH, invN);
}
